// FuzzyMealy_13022340842115
// MI455X (gfx1250) — matrix operand order repaired and checked by reading, not compiled, not run on hardware
//
#include <hip/hip_runtime.h>
#include <hip/hip_bf16.h>

// ---------------------------------------------------------------------------
// FuzzyMealy on MI455X (gfx1250, wave32)
//
//  Phase 0: softmax(T_logits) -> Tb (bf16, [s][i][n] row layout)
//           softmax(O_logits) -> Obw (bf16, pre-swizzled into WMMA-B tiling)
//  Phase 1: sequential state recurrence (64 persistent WGs, column split,
//           LDS reduce, monotonic atomic grid barrier, prefetch of next slice)
//  Phase 2: bucket timesteps by symbol (tiny single-WG kernel)
//  Phase 3: grouped GEMM outputs via v_wmma_f32_16x16x32_bf16 (branch-free:
//           padding rows read from a dedicated zero row, no EXEC games)
// ---------------------------------------------------------------------------

typedef __attribute__((ext_vector_type(16))) __bf16 v16bf;
typedef __attribute__((ext_vector_type(8)))  __bf16 v8bf;
typedef __attribute__((ext_vector_type(8)))  float  v8f;

#define NSTATES 1024
#define NINPUTS 256
#define NOUT    1024
#define SEQLEN  4096
#define RECUR_BLOCKS 64

__device__ __forceinline__ unsigned short f2bf(float f) {
    unsigned int u = __float_as_uint(f);
    unsigned int r = u + 0x7fffu + ((u >> 16) & 1u);   // round-to-nearest-even
    return (unsigned short)(r >> 16);
}
__device__ __forceinline__ float bf2f(unsigned short h) {
    return __uint_as_float(((unsigned int)h) << 16);
}

// ---------------------------------------------------------------------------
// Phase 0: row softmax over last dim (1024). grid.x = 1024*256 rows, grid.y:
// 0 -> T (plain bf16 layout), 1 -> O (swizzled into WMMA-B tile layout):
//   Obw[(((i*32 + (s>>5))*64 + (n>>4))*32 + ((n&15) + 16*((s>>4)&1)))*16 + (s&15)]
// so that for a (symbol, kChunk, nTile) the 32 lanes x 16 bf16 are contiguous.
// ---------------------------------------------------------------------------
__global__ __launch_bounds__(256) void softmax_prep(
    const float* __restrict__ Tl, const float* __restrict__ Ol,
    unsigned short* __restrict__ Tb, unsigned short* __restrict__ Obw) {
    const int row = blockIdx.x;              // row = s*256 + i
    const int tid = threadIdx.x;
    const bool isO = (blockIdx.y != 0);
    const float* src = (isO ? Ol : Tl) + (size_t)row * NSTATES;

    float v0 = src[tid], v1 = src[tid + 256], v2 = src[tid + 512], v3 = src[tid + 768];
    __shared__ float red[256];
    red[tid] = fmaxf(fmaxf(v0, v1), fmaxf(v2, v3));
    __syncthreads();
    for (int st = 128; st > 0; st >>= 1) {
        if (tid < st) red[tid] = fmaxf(red[tid], red[tid + st]);
        __syncthreads();
    }
    const float rowmax = red[0];
    __syncthreads();
    float e0 = __expf(v0 - rowmax), e1 = __expf(v1 - rowmax);
    float e2 = __expf(v2 - rowmax), e3 = __expf(v3 - rowmax);
    red[tid] = e0 + e1 + e2 + e3;
    __syncthreads();
    for (int st = 128; st > 0; st >>= 1) {
        if (tid < st) red[tid] += red[tid + st];
        __syncthreads();
    }
    const float inv = 1.0f / red[0];

    if (!isO) {
        unsigned short* dst = Tb + (size_t)row * NSTATES;
        dst[tid]       = f2bf(e0 * inv);
        dst[tid + 256] = f2bf(e1 * inv);
        dst[tid + 512] = f2bf(e2 * inv);
        dst[tid + 768] = f2bf(e3 * inv);
    } else {
        const int s = row >> 8, i = row & 255;
        const int half = (s >> 4) & 1, kc = s >> 5, e = s & 15;
        const float vals[4] = {e0 * inv, e1 * inv, e2 * inv, e3 * inv};
#pragma unroll
        for (int j = 0; j < 4; ++j) {
            const int n = tid + j * 256;
            const size_t idx =
                ((((size_t)i * 32 + kc) * 64 + (n >> 4)) * 32 +
                 ((n & 15) + 16 * half)) * 16 + e;
            Obw[idx] = f2bf(vals[j]);
        }
    }
}

// ---------------------------------------------------------------------------
// Init: state0 = e0 (fp32 ping-pong + bf16 copy for the output GEMM), bar = 0,
// and the shared zero-row used by padding lanes in the WMMA kernel.
// ---------------------------------------------------------------------------
__global__ __launch_bounds__(1024) void init_kernel(
    float* __restrict__ stpp, unsigned short* __restrict__ sbf,
    unsigned short* __restrict__ zrow, int* __restrict__ bar) {
    const int tid = threadIdx.x;
    const float v = (tid == 0) ? 1.0f : 0.0f;
    stpp[tid] = v;
    stpp[NSTATES + tid] = 0.0f;
    sbf[tid] = f2bf(v);
    zrow[tid] = 0;
    if (tid == 0) bar[0] = 0;
}

// ---------------------------------------------------------------------------
// Phase 1: sequential recurrence. 64 blocks x 256 threads, persistent.
// Block b owns output columns [16b, 16b+16); thread (seg,colIdx) accumulates
// 64 s-values; LDS reduction (fixed order -> bit-deterministic). Grid barrier
// is a monotonic atomic counter. Next step's T slice is prefetched while the
// barrier drains (input sequence is known, so memory pipelines across steps).
// ---------------------------------------------------------------------------
__global__ __launch_bounds__(256) void recur_kernel(
    const int* __restrict__ seq, const unsigned short* __restrict__ Tb,
    volatile float* stpp, unsigned short* __restrict__ sbf, volatile int* bar) {
    const int tid = threadIdx.x;
    const int col = blockIdx.x * 16 + (tid & 15);
    const int seg = tid >> 4;
    const int s0  = seg * 64;
    __shared__ float red[256];

    for (int t = 0; t < SEQLEN; ++t) {
        const int inp = seq[t];
        const volatile float* st = stpp + (t & 1) * NSTATES;
        const unsigned short* Ts = Tb + (size_t)inp * NSTATES + col;  // (s,col) stride 256*1024
        float acc = 0.0f;
#pragma unroll 4
        for (int s = 0; s < 64; ++s) {
            acc += st[s0 + s] * bf2f(Ts[(size_t)(s0 + s) * (NINPUTS * NSTATES)]);
        }
        red[tid] = acc;
        __syncthreads();
        if (seg == 0) {   // tid < 16: final reduction for this block's 16 columns
            float v = 0.0f;
#pragma unroll
            for (int j = 0; j < 16; ++j) v += red[j * 16 + tid];
            stpp[((t + 1) & 1) * NSTATES + col] = v;
            if (t + 1 < SEQLEN) sbf[(size_t)(t + 1) * NSTATES + col] = f2bf(v);
        }
        __threadfence();
        __syncthreads();
        if (tid == 0) atomicAdd((int*)bar, 1);

        // Prefetch next step's T columns while the barrier drains.
        if (t + 1 < SEQLEN) {
            const unsigned short* Tn = Tb + (size_t)seq[t + 1] * NSTATES + col;
#pragma unroll
            for (int j = 0; j < 4; ++j)
                __builtin_prefetch((const void*)&Tn[(size_t)(s0 + j * 16) * (NINPUTS * NSTATES)], 0, 0);
        }
        if (tid == 0) {
            const int target = RECUR_BLOCKS * (t + 1);
            while (*bar < target) __builtin_amdgcn_s_sleep(1);
        }
        __syncthreads();
        __threadfence();
    }
}

// ---------------------------------------------------------------------------
// Phase 2: bucket timesteps by symbol; build padded 16-row tile list.
// Single workgroup; per-group order is irrelevant (each row is independent).
// ---------------------------------------------------------------------------
__global__ __launch_bounds__(256) void group_kernel(
    const int* __restrict__ seq, int* __restrict__ perm,
    int* __restrict__ tileSym, int* __restrict__ tileOff,
    int* __restrict__ tileValid, int* __restrict__ numTiles) {
    __shared__ int cnt[256];
    __shared__ int off[256];
    const int tid = threadIdx.x;
    cnt[tid] = 0;
    __syncthreads();
    for (int t = tid; t < SEQLEN; t += 256) atomicAdd(&cnt[seq[t]], 1);
    __syncthreads();
    if (tid == 0) {
        int run = 0, nt = 0;
        for (int i = 0; i < 256; ++i) {
            off[i] = run;
            const int c = cnt[i];
            for (int j = 0; j < c; j += 16) {
                tileSym[nt]   = i;
                tileOff[nt]   = run + j;
                tileValid[nt] = (c - j < 16) ? (c - j) : 16;
                ++nt;
            }
            run += c;
        }
        numTiles[0] = nt;   // <= 496
    }
    __syncthreads();
    cnt[tid] = 0;
    __syncthreads();
    for (int t = tid; t < SEQLEN; t += 256) {
        const int s = seq[t];
        const int p = off[s] + atomicAdd(&cnt[s], 1);
        perm[p] = t;
    }
}

// ---------------------------------------------------------------------------
// Phase 3: grouped GEMM outputs with v_wmma_f32_16x16x32_bf16.
// grid = (512 tiles, 16 n-groups), 128 threads = 4 waves; wave w handles the
// 16-column tile nt = blockIdx.y*4 + w.  A: bf16 state rows (two contiguous
// 16B chunks per lane per K-chunk); padding lanes read a shared zero row so
// the inner loop is branch-free.  B: pre-swizzled Obw, one contiguous 32B per
// lane -> each wave's B fetch is a single coalesced 1KB block.
// ---------------------------------------------------------------------------
__global__ __launch_bounds__(128) void outgemm_kernel(
    const unsigned short* __restrict__ sbf, const unsigned short* __restrict__ Obw,
    const unsigned short* __restrict__ zrow,
    const int* __restrict__ perm, const int* __restrict__ tileSym,
    const int* __restrict__ tileOff, const int* __restrict__ tileValid,
    const int* __restrict__ numTiles, float* __restrict__ out) {
    const int tile = blockIdx.x;
    if (tile >= numTiles[0]) return;
    const int wave = threadIdx.x >> 5;
    const int lane = threadIdx.x & 31;
    const int sym   = tileSym[tile];
    const int base  = tileOff[tile];
    const int valid = tileValid[tile];
    const int nt = blockIdx.y * 4 + wave;           // 0..63
    const int n0 = nt * 16;
    const int m    = lane & 15;                     // A-row for this lane
    const int half = lane >> 4;
    const bool rowOK = (m < valid);
    // Branch-free padding: lanes beyond the group's row count stream zeros.
    const unsigned short* Arow =
        rowOK ? (sbf + (size_t)perm[base + m] * NSTATES) : zrow;
    const unsigned short* Bbase =
        Obw + (((size_t)sym * 32) * 64 + nt) * 512 + (size_t)lane * 16;

    v8f acc = {};
#pragma unroll
    for (int kc = 0; kc < 32; ++kc) {
        const int k0 = kc * 32;
        const v8bf alo = *(const v8bf*)(Arow + k0 + 16 * half);       // K = k0+16h .. +7
        const v8bf ahi = *(const v8bf*)(Arow + k0 + 16 * half + 8);   // K = k0+16h+8 .. +7
        const unsigned short* Bp = Bbase + (size_t)kc * (64 * 512);
        const v8bf blo = *(const v8bf*)(Bp);
        const v8bf bhi = *(const v8bf*)(Bp + 8);
        v16bf A, B;
#pragma unroll
        for (int e = 0; e < 8; ++e) {
            A[e] = alo[e]; A[e + 8] = ahi[e];
            B[e] = blo[e]; B[e + 8] = bhi[e];
        }
        acc = __builtin_amdgcn_wmma_f32_16x16x32_bf16(
            /*neg_a=*/false, A, /*neg_b=*/false, B,
            /*c_mod=*/(short)0, acc, /*reuse_a=*/false, /*reuse_b=*/false);
    }

    // D layout: VGPR r -> M = r + 8*half, N = lane&15
#pragma unroll
    for (int r = 0; r < 8; ++r) {
        const int mm = r + 8 * half;
        if (mm < valid) {
            const int tr = perm[base + mm];
            out[(size_t)tr * NOUT + n0 + (lane & 15)] = acc[r];
        }
    }
}

// ---------------------------------------------------------------------------
// Host launcher. Workspace layout (needs ~1.03 GB):
//   [0, 512MB)        Tb   bf16 softmax(T), [s][i][n]
//   [512MB, 1024MB)   Obw  bf16 softmax(O), WMMA-B swizzled
//   [1024MB, +8MB)    sbf  bf16 states [4096][1024]
//   +8MB              stpp fp32 ping-pong state [2][1024]  (8KB)
//   +8MB+8KB          zrow bf16 zero row [1024]            (2KB)
//   +8MB+64KB         misc: bar, numTiles, perm[4096], tileSym/Off/Valid[512]
// ---------------------------------------------------------------------------
extern "C" void kernel_launch(void* const* d_in, const int* in_sizes, int n_in,
                              void* d_out, int out_size, void* d_ws, size_t ws_size,
                              hipStream_t stream) {
    (void)in_sizes; (void)n_in; (void)out_size; (void)ws_size;
    const int*   seq = (const int*)d_in[0];
    const float* Tl  = (const float*)d_in[1];
    const float* Ol  = (const float*)d_in[2];
    float* out = (float*)d_out;

    char* ws = (char*)d_ws;
    unsigned short* Tb  = (unsigned short*)(ws);
    unsigned short* Obw = (unsigned short*)(ws + ((size_t)512 << 20));
    unsigned short* sbf = (unsigned short*)(ws + ((size_t)1024 << 20));
    char* tail          = ws + ((size_t)1024 << 20) + ((size_t)8 << 20);
    float* stpp         = (float*)(tail);
    unsigned short* zrow = (unsigned short*)(tail + 8192);
    int* misc           = (int*)(tail + 65536);
    int* bar       = misc;
    int* numTiles  = misc + 64;
    int* perm      = misc + 128;
    int* tileSym   = perm + SEQLEN;
    int* tileOff   = tileSym + 512;
    int* tileValid = tileOff + 512;

    softmax_prep<<<dim3(NSTATES * NINPUTS, 2), 256, 0, stream>>>(Tl, Ol, Tb, Obw);
    init_kernel<<<1, 1024, 0, stream>>>(stpp, sbf, zrow, bar);
    group_kernel<<<1, 256, 0, stream>>>(seq, perm, tileSym, tileOff, tileValid, numTiles);
    recur_kernel<<<RECUR_BLOCKS, 256, 0, stream>>>(seq, Tb, stpp, sbf, bar);
    outgemm_kernel<<<dim3(512, 16), 128, 0, stream>>>(sbf, Obw, zrow, perm, tileSym,
                                                      tileOff, tileValid, numTiles, out);
}
